// CausalSelfAttention_36893769072969
// MI455X (gfx1250) — matrix operand order repaired and checked by reading, not compiled, not run on hardware
//
#include <hip/hip_runtime.h>

// ---------------------------------------------------------------------------
// CDNA5 (gfx1250, wave32) causal self-attention:
//   qkv proj (bf16 WMMA, 64x64 wave tiles) -> flash attention (bf16 WMMA,
//   online softmax) -> out proj (bf16 WMMA, f32 out)
// ---------------------------------------------------------------------------

typedef __attribute__((ext_vector_type(16))) __bf16 bf16x16;
typedef __attribute__((ext_vector_type(8)))  float  f32x8;
typedef unsigned int u32;
typedef unsigned short u16;
typedef __attribute__((ext_vector_type(4))) u32 u32x4;
typedef __attribute__((ext_vector_type(2))) u32 u32x2;
typedef __attribute__((ext_vector_type(4))) float f32x4;

#define N_EMBD   1024
#define N_HEAD   16
#define HEAD_DIM 64
#define TSEQ     2048
#define BATCH    2
#define QK_SCALE 0.125f  // 1/sqrt(64)

union FragU { bf16x16 v; u32x4 q[2]; };

static __device__ __forceinline__ u16 f2bf(float x) {
  union { float f; u32 u; } c; c.f = x;
  u32 u = c.u;
  u32 r = u + 0x7FFFu + ((u >> 16) & 1u);   // round-to-nearest-even
  return (u16)(r >> 16);
}

// A-matrix 16x32 bf16 fragment (ISA 7.12.2): lane<16 holds row M=lane,
// K = [0..7] in V0..3 and [16..23] in V4..7; lane>=16 holds K=[8..15],[24..31].
static __device__ __forceinline__ bf16x16 load_afrag(const u16* base, int ld,
                                                     int row0, int k0, int lane) {
  int half = lane >> 4, r = lane & 15;
  const u16* p = base + (size_t)(row0 + r) * ld + k0 + half * 8;
  FragU f;
  f.q[0] = *(const u32x4*)p;
  f.q[1] = *(const u32x4*)(p + 16);
  return f.v;
}

// B-matrix 32x16 bf16 fragment: lane<16 holds column N=lane, K=[0..7],[16..23]
// (2/VGPR); lane>=16 holds K=[8..15],[24..31]. Source array is row-major [N][K]
// (i.e. W stored N x K, or V stored transposed D x T).
static __device__ __forceinline__ bf16x16 load_bfrag(const u16* base, int ld,
                                                     int n0, int k0, int lane) {
  int half = lane >> 4, c = lane & 15;
  const u16* p = base + (size_t)(n0 + c) * ld + k0 + half * 8;
  FragU f;
  f.q[0] = *(const u32x4*)p;
  f.q[1] = *(const u32x4*)(p + 16);
  return f.v;
}

static __device__ __forceinline__ f32x8 wmma_bf16(bf16x16 a, bf16x16 b, f32x8 c) {
  return __builtin_amdgcn_wmma_f32_16x16x32_bf16(false, a, false, b,
                                                 (short)0, c, false, false);
}

// ---------------------------------------------------------------------------
// Vectorized f32 -> bf16 conversion: 4 elements / thread (b128 in, b64 out).
__global__ void cvt_f32_bf16(const float* __restrict__ src, u16* __restrict__ dst, int n4) {
  int i = blockIdx.x * blockDim.x + threadIdx.x;
  if (i < n4) {
    f32x4 v = *(const f32x4*)(src + (size_t)i * 4);
    u32x2 o;
    o.x = (u32)f2bf(v.x) | ((u32)f2bf(v.y) << 16);
    o.y = (u32)f2bf(v.z) | ((u32)f2bf(v.w) << 16);
    *(u32x2*)(dst + (size_t)i * 4) = o;
  }
}

// ---------------------------------------------------------------------------
// QKV projection: out = x @ W^T + b.  z selects Q/K/V.
// Q,K stored (B,H,T,D) bf16 (Q pre-scaled); V stored (B,H,D,T) bf16.
// Block = 128 thr (4 waves, 2x2); wave tile 64x64 (16 accums -> 4x fragment
// reuse, halving L2 fragment traffic vs a 32x32 tile); block tile 128x128.
__global__ __launch_bounds__(128) void qkv_proj(
    const u16* __restrict__ xb,
    const u16* __restrict__ wqb, const u16* __restrict__ wkb, const u16* __restrict__ wvb,
    const float* __restrict__ bq, const float* __restrict__ bk, const float* __restrict__ bv,
    u16* __restrict__ qo, u16* __restrict__ ko, u16* __restrict__ vto) {
  int lane = threadIdx.x & 31;
  int wave = threadIdx.x >> 5;
  int M0 = blockIdx.x * 128 + (wave & 1) * 64;
  int N0 = blockIdx.y * 128 + (wave >> 1) * 64;
  int z = blockIdx.z;
  const u16*   w    = (z == 0) ? wqb : (z == 1) ? wkb : wvb;
  const float* bias = (z == 0) ? bq : (z == 1) ? bk : bv;

  f32x8 acc[4][4];
#pragma unroll
  for (int i = 0; i < 4; ++i)
#pragma unroll
    for (int j = 0; j < 4; ++j)
      acc[i][j] = (f32x8){0.f,0.f,0.f,0.f,0.f,0.f,0.f,0.f};

  for (int k0 = 0; k0 < N_EMBD; k0 += 32) {
    if (k0 + 32 < N_EMBD) {
      __builtin_prefetch(xb + (size_t)(M0 + (lane & 15)) * N_EMBD + k0 + 32, 0, 1);
      __builtin_prefetch(w  + (size_t)(N0 + (lane & 15)) * N_EMBD + k0 + 32, 0, 1);
    }
    bf16x16 a[4];
#pragma unroll
    for (int mi = 0; mi < 4; ++mi)
      a[mi] = load_afrag(xb, N_EMBD, M0 + mi * 16, k0, lane);
#pragma unroll
    for (int ni = 0; ni < 4; ++ni) {
      bf16x16 bfr = load_bfrag(w, N_EMBD, N0 + ni * 16, k0, lane);
#pragma unroll
      for (int mi = 0; mi < 4; ++mi)
        acc[mi][ni] = wmma_bf16(a[mi], bfr, acc[mi][ni]);
    }
  }

  int half = lane >> 4, c = lane & 15;
#pragma unroll
  for (int ni = 0; ni < 4; ++ni) {
    int n = N0 + ni * 16 + c;
    float bval = bias[n];
    int h = n >> 6, d = n & 63;
#pragma unroll
    for (int mi = 0; mi < 4; ++mi) {
#pragma unroll
      for (int v = 0; v < 8; ++v) {
        int gm = M0 + mi * 16 + half * 8 + v;      // global row = b*T + t
        int b  = gm >> 11;
        int t  = gm & (TSEQ - 1);
        float val = acc[mi][ni][v] + bval;
        if (z == 0)
          qo[(((size_t)(b * N_HEAD + h)) * TSEQ + t) * HEAD_DIM + d] = f2bf(val * QK_SCALE);
        else if (z == 1)
          ko[(((size_t)(b * N_HEAD + h)) * TSEQ + t) * HEAD_DIM + d] = f2bf(val);
        else
          vto[(((size_t)(b * N_HEAD + h)) * HEAD_DIM + d) * TSEQ + t] = f2bf(val);
      }
    }
  }
}

// ---------------------------------------------------------------------------
// Flash attention. 1 wave per block; block handles 32 query rows of one (b,h).
// S tile: 32(q) x 64(k) via wmma; online softmax fully lane-aligned in the
// C/D register layout; P transposed D-layout -> A-layout through LDS.
__global__ __launch_bounds__(32) void attn(
    const u16* __restrict__ qb, const u16* __restrict__ kb,
    const u16* __restrict__ vtb, u16* __restrict__ ob) {
  __shared__ u16 lds_p[32 * 72];   // 32 rows, pitch 72 (16B-aligned rows)
  int lane = threadIdx.x & 31;
  int half = lane >> 4, c = lane & 15;
  int Q0 = blockIdx.x * 32;
  int h  = blockIdx.y;
  int b  = blockIdx.z;

  const u16* qbase = qb  + ((size_t)(b * N_HEAD + h)) * TSEQ * HEAD_DIM;
  const u16* kbase = kb  + ((size_t)(b * N_HEAD + h)) * TSEQ * HEAD_DIM;
  const u16* vbase = vtb + ((size_t)(b * N_HEAD + h)) * HEAD_DIM * TSEQ;

  bf16x16 qa[2][2];
#pragma unroll
  for (int mi = 0; mi < 2; ++mi)
#pragma unroll
    for (int kc = 0; kc < 2; ++kc)
      qa[mi][kc] = load_afrag(qbase, HEAD_DIM, Q0 + mi * 16, kc * 32, lane);

  f32x8 of[2][4];
  float mrun[2][8], lrun[2][8];
#pragma unroll
  for (int mi = 0; mi < 2; ++mi) {
#pragma unroll
    for (int di = 0; di < 4; ++di)
      of[mi][di] = (f32x8){0.f,0.f,0.f,0.f,0.f,0.f,0.f,0.f};
#pragma unroll
    for (int v = 0; v < 8; ++v) { mrun[mi][v] = -1e30f; lrun[mi][v] = 0.f; }
  }

  int nkt = ((Q0 + 31) >> 6) + 1;          // 64-wide key tiles, causal limit
  for (int kt = 0; kt < nkt; ++kt) {
    int K0 = kt * 64;

    // ---- S = Q @ K^T (pre-scaled Q) ----
    f32x8 s[2][4];
#pragma unroll
    for (int ni = 0; ni < 4; ++ni) {
      bf16x16 kf0 = load_bfrag(kbase, HEAD_DIM, K0 + ni * 16, 0,  lane);
      bf16x16 kf1 = load_bfrag(kbase, HEAD_DIM, K0 + ni * 16, 32, lane);
#pragma unroll
      for (int mi = 0; mi < 2; ++mi) {
        f32x8 t0 = (f32x8){0.f,0.f,0.f,0.f,0.f,0.f,0.f,0.f};
        t0 = wmma_bf16(qa[mi][0], kf0, t0);
        t0 = wmma_bf16(qa[mi][1], kf1, t0);
        s[mi][ni] = t0;
      }
    }

    // ---- causal mask (boundary tiles only) ----
    if (K0 + 63 > Q0) {
#pragma unroll
      for (int mi = 0; mi < 2; ++mi)
#pragma unroll
        for (int ni = 0; ni < 4; ++ni)
#pragma unroll
          for (int v = 0; v < 8; ++v) {
            int qg = Q0 + mi * 16 + half * 8 + v;
            int kg = K0 + ni * 16 + c;
            if (kg > qg) s[mi][ni][v] = -1e30f;
          }
    }

    // ---- online softmax (row data lane-aligned in D layout) ----
#pragma unroll
    for (int mi = 0; mi < 2; ++mi) {
#pragma unroll
      for (int v = 0; v < 8; ++v) {
        float tm = fmaxf(fmaxf(s[mi][0][v], s[mi][1][v]),
                         fmaxf(s[mi][2][v], s[mi][3][v]));
        tm = fmaxf(tm, __shfl_xor(tm, 1, 32));
        tm = fmaxf(tm, __shfl_xor(tm, 2, 32));
        tm = fmaxf(tm, __shfl_xor(tm, 4, 32));
        tm = fmaxf(tm, __shfl_xor(tm, 8, 32));
        float mn = fmaxf(mrun[mi][v], tm);
        float alpha = __expf(mrun[mi][v] - mn);
        mrun[mi][v] = mn;
        float rs = 0.f;
#pragma unroll
        for (int ni = 0; ni < 4; ++ni) {
          float p = __expf(s[mi][ni][v] - mn);
          s[mi][ni][v] = p;
          rs += p;
        }
        rs += __shfl_xor(rs, 1, 32);
        rs += __shfl_xor(rs, 2, 32);
        rs += __shfl_xor(rs, 4, 32);
        rs += __shfl_xor(rs, 8, 32);
        lrun[mi][v] = lrun[mi][v] * alpha + rs;
#pragma unroll
        for (int di = 0; di < 4; ++di) of[mi][di][v] *= alpha;
      }
    }

    // ---- P: D-layout regs -> LDS row-major (implicit transpose) ----
#pragma unroll
    for (int mi = 0; mi < 2; ++mi)
#pragma unroll
      for (int ni = 0; ni < 4; ++ni)
#pragma unroll
        for (int v = 0; v < 8; ++v) {
          int row = mi * 16 + half * 8 + v;
          lds_p[row * 72 + ni * 16 + c] = f2bf(s[mi][ni][v]);
        }
    __syncthreads();

    bf16x16 pa[2][2];
#pragma unroll
    for (int mi = 0; mi < 2; ++mi)
#pragma unroll
      for (int kc = 0; kc < 2; ++kc)
        pa[mi][kc] = load_afrag(lds_p, 72, mi * 16, kc * 32, lane);
    __syncthreads();

    // ---- O += P @ V  (V stored transposed: rows of vt = columns of V) ----
#pragma unroll
    for (int di = 0; di < 4; ++di) {
      bf16x16 vf0 = load_bfrag(vbase, TSEQ, di * 16, K0,      lane);
      bf16x16 vf1 = load_bfrag(vbase, TSEQ, di * 16, K0 + 32, lane);
#pragma unroll
      for (int mi = 0; mi < 2; ++mi) {
        of[mi][di] = wmma_bf16(pa[mi][0], vf0, of[mi][di]);
        of[mi][di] = wmma_bf16(pa[mi][1], vf1, of[mi][di]);
      }
    }
  }

  // ---- normalize and store O as bf16 (B,T,C) ----
#pragma unroll
  for (int mi = 0; mi < 2; ++mi)
#pragma unroll
    for (int v = 0; v < 8; ++v) {
      float inv = 1.f / lrun[mi][v];
      int t = Q0 + mi * 16 + half * 8 + v;
#pragma unroll
      for (int di = 0; di < 4; ++di) {
        int col = h * HEAD_DIM + di * 16 + c;
        ob[((size_t)(b * TSEQ) + t) * N_EMBD + col] = f2bf(of[mi][di][v] * inv);
      }
    }
}

// ---------------------------------------------------------------------------
// Output projection: out = O @ Wo^T + bo, f32 output. Same 64x64 wave tiling.
__global__ __launch_bounds__(128) void out_proj(
    const u16* __restrict__ ab, const u16* __restrict__ wob,
    const float* __restrict__ bo, float* __restrict__ out) {
  int lane = threadIdx.x & 31;
  int wave = threadIdx.x >> 5;
  int M0 = blockIdx.x * 128 + (wave & 1) * 64;
  int N0 = blockIdx.y * 128 + (wave >> 1) * 64;

  f32x8 acc[4][4];
#pragma unroll
  for (int i = 0; i < 4; ++i)
#pragma unroll
    for (int j = 0; j < 4; ++j)
      acc[i][j] = (f32x8){0.f,0.f,0.f,0.f,0.f,0.f,0.f,0.f};

  for (int k0 = 0; k0 < N_EMBD; k0 += 32) {
    if (k0 + 32 < N_EMBD) {
      __builtin_prefetch(ab  + (size_t)(M0 + (lane & 15)) * N_EMBD + k0 + 32, 0, 1);
      __builtin_prefetch(wob + (size_t)(N0 + (lane & 15)) * N_EMBD + k0 + 32, 0, 1);
    }
    bf16x16 a[4];
#pragma unroll
    for (int mi = 0; mi < 4; ++mi)
      a[mi] = load_afrag(ab, N_EMBD, M0 + mi * 16, k0, lane);
#pragma unroll
    for (int ni = 0; ni < 4; ++ni) {
      bf16x16 bfr = load_bfrag(wob, N_EMBD, N0 + ni * 16, k0, lane);
#pragma unroll
      for (int mi = 0; mi < 4; ++mi)
        acc[mi][ni] = wmma_bf16(a[mi], bfr, acc[mi][ni]);
    }
  }

  int half = lane >> 4, c = lane & 15;
#pragma unroll
  for (int ni = 0; ni < 4; ++ni) {
    int n = N0 + ni * 16 + c;
    float bval = bo[n];
#pragma unroll
    for (int mi = 0; mi < 4; ++mi)
#pragma unroll
      for (int v = 0; v < 8; ++v) {
        int gm = M0 + mi * 16 + half * 8 + v;
        out[(size_t)gm * N_EMBD + n] = acc[mi][ni][v] + bval;
      }
  }
}

// ---------------------------------------------------------------------------
extern "C" void kernel_launch(void* const* d_in, const int* in_sizes, int n_in,
                              void* d_out, int out_size, void* d_ws, size_t ws_size,
                              hipStream_t stream) {
  const float* x  = (const float*)d_in[0];
  const float* Wq = (const float*)d_in[1];
  const float* bq = (const float*)d_in[2];
  const float* Wk = (const float*)d_in[3];
  const float* bk = (const float*)d_in[4];
  const float* Wv = (const float*)d_in[5];
  const float* bv = (const float*)d_in[6];
  const float* Wo = (const float*)d_in[7];
  const float* bo = (const float*)d_in[8];
  float* out = (float*)d_out;

  const size_t XE = (size_t)BATCH * TSEQ * N_EMBD;  // 4,194,304
  const size_t WE = (size_t)N_EMBD * N_EMBD;        // 1,048,576

  u16* xb  = (u16*)d_ws;
  u16* wqb = xb  + XE;
  u16* wkb = wqb + WE;
  u16* wvb = wkb + WE;
  u16* wob = wvb + WE;
  u16* qbf = wob + WE;
  u16* kbf = qbf + XE;
  u16* vtb = kbf + XE;
  u16* obf = vtb + XE;   // total ~50 MB of workspace

  cvt_f32_bf16<<<dim3((u32)(XE / 4 + 255) / 256), dim3(256), 0, stream>>>(x,  xb,  (int)(XE / 4));
  cvt_f32_bf16<<<dim3((u32)(WE / 4 + 255) / 256), dim3(256), 0, stream>>>(Wq, wqb, (int)(WE / 4));
  cvt_f32_bf16<<<dim3((u32)(WE / 4 + 255) / 256), dim3(256), 0, stream>>>(Wk, wkb, (int)(WE / 4));
  cvt_f32_bf16<<<dim3((u32)(WE / 4 + 255) / 256), dim3(256), 0, stream>>>(Wv, wvb, (int)(WE / 4));
  cvt_f32_bf16<<<dim3((u32)(WE / 4 + 255) / 256), dim3(256), 0, stream>>>(Wo, wob, (int)(WE / 4));

  // M = B*T = 4096 -> 32 tiles of 128; N = 1024 -> 8 tiles of 128; z = q/k/v
  qkv_proj<<<dim3(32, 8, 3), dim3(128), 0, stream>>>(
      xb, wqb, wkb, wvb, bq, bk, bv, qbf, kbf, vtb);

  attn<<<dim3(TSEQ / 32, N_HEAD, BATCH), dim3(32), 0, stream>>>(qbf, kbf, vtb, obf);

  out_proj<<<dim3(32, 8), dim3(128), 0, stream>>>(obf, wob, bo, out);
}
